// ScaledDotProductAttention_8126078124365
// MI455X (gfx1250) — matrix operand order repaired and checked by reading, not compiled, not run on hardware
//
#include <hip/hip_runtime.h>
#include <hip/hip_bf16.h>
#include <math.h>

typedef __attribute__((ext_vector_type(16))) _Float16     v16h;
typedef __attribute__((ext_vector_type(8)))  float        v8f;
typedef __attribute__((ext_vector_type(4)))  float        f32x4;
typedef __attribute__((ext_vector_type(4)))  unsigned int u32x4;

#define S_LEN   4096
#define DHEAD   64
#define BLOCK_M 64
#define NWAVES  4
#define TILE_K  32

#define LDK   72   // halves per K-tile row   (64 + 8 pad)  -> 144 B, 16B aligned runs
#define LDV   40   // halves per V^T-tile row (32 + 8 pad)  ->  80 B, 16B aligned runs
#define LDS_S 36   // floats per score row    (32 + 4 pad)  -> 144 B, avoids bank clash

union HF {
    v16h h;
    u32x4 q[2];
    unsigned int u[8];
};

static __device__ __forceinline__ unsigned int pkh2(float lo, float hi) {
    // v_cvt_pk_rtz_f16_f32: pack two f32 -> f16x2 in one VALU op
    return __builtin_bit_cast(unsigned int, __builtin_amdgcn_cvt_pkrtz(lo, hi));
}

static __device__ __forceinline__ float exp2_fast(float x) {
    // raw v_exp_f32; args are <= 0 here, flush-to-zero on deep negatives is fine
    return __builtin_amdgcn_exp2f(x);
}

__global__ __launch_bounds__(NWAVES * 32)
void fa_causal_wmma_kernel(const float* __restrict__ Q,
                           const float* __restrict__ K,
                           const float* __restrict__ V,
                           float* __restrict__ O) {
    __shared__ __attribute__((aligned(16))) unsigned int sK[TILE_K * (LDK / 2)]; // 4608 B, f16 [k][d]
    __shared__ __attribute__((aligned(16))) unsigned int sV[DHEAD * (LDV / 2)];  // 5120 B, f16 [d][k]
    __shared__ __attribute__((aligned(16))) float sS[NWAVES][16 * LDS_S];        // 9216 B, f32 scores

    const int tid  = threadIdx.x;
    const int lane = tid & 31;
    const int wave = tid >> 5;
    const int b    = blockIdx.y;
    const int qb   = blockIdx.x * BLOCK_M;

    const int lmod   = lane & 15;
    const int lhi    = lane >> 4;     // which half of the wave
    const int base8  = lhi * 8;       // A-matrix K-group offset
    const int base16 = lhi * 16;      // unused; B fragments also use base8

    const float NEG_INF = -__builtin_inff();

    // Uniform per-batch base pointers; all loop addressing is base + 32-bit offset (GVS)
    const float* Kb = K + (size_t)b * S_LEN * DHEAD;
    const float* Vb = V + (size_t)b * S_LEN * DHEAD;

    // Loop-invariant per-thread staging offsets (in floats)
    int kofs[4], vofs[2];
#pragma unroll
    for (int it = 0; it < 4; ++it) {
        int t = tid + 128 * it;                       // 512 float4 tasks
        kofs[it] = (t >> 4) * DHEAD + (t & 15) * 4;   // row (t>>4), dquad
    }
#pragma unroll
    for (int it = 0; it < 2; ++it) {
        int t = tid + 128 * it;                       // 256 tasks: (k-pair, d-quad)
        vofs[it] = ((t & 15) * 2) * DHEAD + (t >> 4) * 4;
    }
    const int pofs = (tid >> 4) * DHEAD + (tid & 15) * 4;  // prefetch offset

    // ------- Q A-fragments (16x32 f16); fold log2(e)/sqrt(D) so softmax runs in base-2 -------
    const int qrow = qb + wave * 16 + lmod;
    const float* qp = Q + ((size_t)(b * S_LEN + qrow)) * DHEAD;
    const float sc = 0.125f * 1.44269504088896340736f; // log2(e)/sqrt(64)
    HF qf[2];
#pragma unroll
    for (int f = 0; f < 2; ++f) {
        const int d0 = f * 32 + base8;            // two 8-elem runs: d0.. , d0+16..
        f32x4 r0 = *(const f32x4*)(qp + d0);
        f32x4 r1 = *(const f32x4*)(qp + d0 + 4);
        f32x4 r2 = *(const f32x4*)(qp + d0 + 16);
        f32x4 r3 = *(const f32x4*)(qp + d0 + 20);
        qf[f].u[0] = pkh2(r0.x * sc, r0.y * sc);
        qf[f].u[1] = pkh2(r0.z * sc, r0.w * sc);
        qf[f].u[2] = pkh2(r1.x * sc, r1.y * sc);
        qf[f].u[3] = pkh2(r1.z * sc, r1.w * sc);
        qf[f].u[4] = pkh2(r2.x * sc, r2.y * sc);
        qf[f].u[5] = pkh2(r2.z * sc, r2.w * sc);
        qf[f].u[6] = pkh2(r3.x * sc, r3.y * sc);
        qf[f].u[7] = pkh2(r3.z * sc, r3.w * sc);
    }

    v8f acc[4];
    const v8f zero8 = {0.f, 0.f, 0.f, 0.f, 0.f, 0.f, 0.f, 0.f};
#pragma unroll
    for (int t = 0; t < 4; ++t) acc[t] = zero8;

    float m_run = NEG_INF;   // running row max in base-2 domain (row = lmod, both halves)
    float l_run = 0.0f;      // running row sum

    const int qi = qrow;                       // this lane's query row
    const int qwave_min = qb + wave * 16;      // first query row of this wave
    const int qwave_max = qwave_min + 15;      // last query row of this wave
    const int nkt = (qb >> 5) + 2;             // causal: only tiles with kbase <= qb+32

    for (int kt = 0; kt < nkt; ++kt) {
        const int kbase = kt * TILE_K;
        const int koff  = kbase * DHEAD;       // scalar per-iteration term

        // ---- cooperative stage: K tile f32 -> f16 row-major [k][d] ----
#pragma unroll
        for (int it = 0; it < 4; ++it) {
            f32x4 v4 = *(const f32x4*)(Kb + koff + kofs[it]);
            int t  = tid + 128 * it;
            int u0 = (t >> 4) * (LDK / 2) + ((t & 15) * 4 >> 1);
            sK[u0]     = pkh2(v4.x, v4.y);
            sK[u0 + 1] = pkh2(v4.z, v4.w);
        }
        // ---- cooperative stage: V tile f32 -> f16 transposed [d][k] ----
#pragma unroll
        for (int it = 0; it < 2; ++it) {
            f32x4 a = *(const f32x4*)(Vb + koff + vofs[it]);
            f32x4 c = *(const f32x4*)(Vb + koff + vofs[it] + DHEAD);
            int t  = tid + 128 * it;
            int k2 = (t & 15) * 2;
            int d0 = (t >> 4) * 4;
            int kb2 = k2 >> 1;
            sV[(d0 + 0) * (LDV / 2) + kb2] = pkh2(a.x, c.x);
            sV[(d0 + 1) * (LDV / 2) + kb2] = pkh2(a.y, c.y);
            sV[(d0 + 2) * (LDV / 2) + kb2] = pkh2(a.z, c.z);
            sV[(d0 + 3) * (LDV / 2) + kb2] = pkh2(a.w, c.w);
        }

        // ---- prefetch next K/V tile into cache (global_prefetch_b8) ----
        if (kt + 1 < nkt) {
            const int noff = koff + TILE_K * DHEAD + pofs;
            __builtin_prefetch(Kb + noff, 0, 3);
            __builtin_prefetch(Kb + noff + 8 * DHEAD, 0, 3);
            __builtin_prefetch(Kb + noff + 16 * DHEAD, 0, 3);
            __builtin_prefetch(Kb + noff + 24 * DHEAD, 0, 3);
            __builtin_prefetch(Vb + noff, 0, 3);
            __builtin_prefetch(Vb + noff + 16 * DHEAD, 0, 3);
        }
        __syncthreads();

        if (kbase <= qwave_max) {   // wave-uniform: EXEC stays all-1s inside
            // ---------------- S = Q K^T : two 16x16 fragments ----------------
            v8f sfrag[2];
#pragma unroll
            for (int g = 0; g < 2; ++g) {
                const int krow = g * 16 + lmod;   // B-matrix N = key
                const char* kRow = (const char*)sK + (size_t)krow * LDK * 2;
                HF kb0, kb1;
                kb0.q[0] = *(const u32x4*)(kRow + 2 * (0 + base8));
                kb0.q[1] = *(const u32x4*)(kRow + 2 * (0 + base8 + 16));
                kb1.q[0] = *(const u32x4*)(kRow + 2 * (32 + base8));
                kb1.q[1] = *(const u32x4*)(kRow + 2 * (32 + base8 + 16));
                v8f z = zero8;
                z = __builtin_amdgcn_wmma_f32_16x16x32_f16(false, qf[0].h, false, kb0.h,
                                                           (short)0, z, false, false);
                z = __builtin_amdgcn_wmma_f32_16x16x32_f16(false, qf[1].h, false, kb1.h,
                                                           (short)0, z, false, false);
                sfrag[g] = z;
            }
            // ---- transpose scores via per-wave LDS (same-wave LDS is in-order) ----
            float* srow = sS[wave];
#pragma unroll
            for (int g = 0; g < 2; ++g)
#pragma unroll
                for (int r = 0; r < 8; ++r)
                    srow[(r + 8 * lhi) * LDS_S + g * 16 + lmod] = sfrag[g][r];

            const int colbase = lhi * 16;
            float s[16];
#pragma unroll
            for (int j = 0; j < 4; ++j) {
                f32x4 v4 = *(const f32x4*)&srow[lmod * LDS_S + colbase + 4 * j];
                s[4 * j + 0] = v4.x; s[4 * j + 1] = v4.y;
                s[4 * j + 2] = v4.z; s[4 * j + 3] = v4.w;
            }
            // ---- causal mask only on boundary tiles (wave-uniform branch) ----
            if (kbase + TILE_K - 1 > qwave_min) {
#pragma unroll
                for (int j = 0; j < 16; ++j) {
                    int kj = kbase + colbase + j;
                    s[j] = (kj <= qi) ? s[j] : NEG_INF;
                }
            }
            // ---- online softmax update (base-2 domain) ----
            float mt = NEG_INF;
#pragma unroll
            for (int j = 0; j < 16; ++j) mt = fmaxf(mt, s[j]);
            mt = fmaxf(mt, __shfl_xor(mt, 16));
            float mnew = fmaxf(m_run, mt);
            float corr = exp2_fast(m_run - mnew);
            m_run = mnew;
            float p[16];
            float psum = 0.0f;
#pragma unroll
            for (int j = 0; j < 16; ++j) {
                p[j] = exp2_fast(s[j] - mnew);
                psum += p[j];
            }
            psum += __shfl_xor(psum, 16);
            l_run = l_run * corr + psum;

            // ---- rebuild P as f16 A-fragment purely in registers ----
            unsigned int pp[8];
#pragma unroll
            for (int j = 0; j < 8; ++j) pp[j] = pkh2(p[2 * j], p[2 * j + 1]);
            HF pf;
#pragma unroll
            for (int j = 0; j < 4; ++j) {
                unsigned int xlo = (unsigned int)__shfl_xor((int)pp[j], 16);     // partner lo pairs
                unsigned int xhi = (unsigned int)__shfl_xor((int)pp[j + 4], 16); // partner hi pairs
                pf.u[j]     = (lhi == 0) ? pp[j]   : xhi;       // K pairs base8 + {0..7}
                pf.u[j + 4] = (lhi == 0) ? xlo     : pp[j + 4]; // K pairs base8+16 + {0..7}
            }

            // ---- rescale accumulators by 2^(m_old - m_new), broadcast per C-row ----
            float fac[8];
#pragma unroll
            for (int r = 0; r < 8; ++r) fac[r] = __shfl(corr, r + base8);
#pragma unroll
            for (int t = 0; t < 4; ++t)
#pragma unroll
                for (int r = 0; r < 8; ++r) acc[t][r] *= fac[r];

            // ---------------- O += P V ----------------
#pragma unroll
            for (int t = 0; t < 4; ++t) {
                const int dcol = t * 16 + lmod;   // B-matrix N = head dim
                const char* vRow = (const char*)sV + (size_t)dcol * LDV * 2;
                HF vb;
                vb.q[0] = *(const u32x4*)(vRow + 2 * (base8));
                vb.q[1] = *(const u32x4*)(vRow + 2 * (base8 + 16));
                acc[t] = __builtin_amdgcn_wmma_f32_16x16x32_f16(false, pf.h, false, vb.h,
                                                                (short)0, acc[t], false, false);
            }
        }
        __syncthreads();
    }

    // ---------------- epilogue: divide by row sum, store f32 ----------------
    float linv = 1.0f / l_run;
    float fac[8];
#pragma unroll
    for (int r = 0; r < 8; ++r) fac[r] = __shfl(linv, r + base8);
    float* op = O + ((size_t)(b * S_LEN + qb + wave * 16 + 8 * lhi)) * DHEAD + lmod;
#pragma unroll
    for (int t = 0; t < 4; ++t)
#pragma unroll
        for (int r = 0; r < 8; ++r)
            op[(size_t)r * DHEAD + t * 16] = acc[t][r] * fac[r];
}

extern "C" void kernel_launch(void* const* d_in, const int* in_sizes, int n_in,
                              void* d_out, int out_size, void* d_ws, size_t ws_size,
                              hipStream_t stream) {
    (void)n_in; (void)out_size; (void)d_ws; (void)ws_size;
    const float* Q = (const float*)d_in[0];
    const float* K = (const float*)d_in[1];
    const float* V = (const float*)d_in[2];
    float* O = (float*)d_out;
    const int B = in_sizes[0] / (S_LEN * DHEAD);   // 16
    dim3 grid(S_LEN / BLOCK_M, B);
    dim3 block(NWAVES * 32);
    hipLaunchKernelGGL(fa_causal_wmma_kernel, grid, block, 0, stream, Q, K, V, O);
}
